// PointNetSetAbstraction_87273735455096
// MI455X (gfx1250) — matrix operand order repaired and checked by reading, not compiled, not run on hardware
//
#include <hip/hip_runtime.h>
#include <hip/hip_bf16.h>

// ---------------------------------------------------------------------------
// PointNet++ Set Abstraction for MI455X (gfx1250, wave32, WMMA).
// B=16, N=4096, S=1024, K=32, D_FEAT=64, MLP 67->64->64->128, BN over (B,S,K).
// Memory-bound pipeline => bf16 activations, f32 WMMA accumulation, fused
// BN/ReLU into next layer's A-fragment load, fused BN statistics in epilogue.
// ---------------------------------------------------------------------------

#define BB   16
#define NN   4096
#define SS   1024
#define KK   32
#define DF   64
#define MROWS (BB * SS * KK)      // 524288 GEMM rows
#define C0PAD 96                  // 67 channels padded to 3 K-chunks of 32

typedef __attribute__((ext_vector_type(16))) __bf16 v16bf;
typedef __attribute__((ext_vector_type(8)))  __bf16 v8bf;
typedef __attribute__((ext_vector_type(8)))  float  v8f;

// ------------------------------- utility -----------------------------------
__global__ void zero_f32_kernel(float* __restrict__ p, int n) {
  int i = blockIdx.x * blockDim.x + threadIdx.x;
  if (i < n) p[i] = 0.0f;
}

// Convert weights (Cout x Cin, f32) to bf16, K-padded to Cpad with zeros.
__global__ void prep_w_kernel(const float* __restrict__ W, __bf16* __restrict__ out,
                              int Cout, int Cin, int Cpad) {
  int t = blockIdx.x * blockDim.x + threadIdx.x;
  if (t >= Cout * Cpad) return;
  int o = t / Cpad, c = t % Cpad;
  out[t] = (c < Cin) ? (__bf16)W[o * Cin + c] : (__bf16)0.0f;
}

// ------------------------------- KNN ---------------------------------------
// One wave32 per query point. xyz[b] tile cached in LDS. 32 extraction passes:
// lexicographic (dist,idx) min over previously-unselected keys, wave-reduced
// with __shfl_xor. Matches top_k tie-breaking (lower index wins on equal dist).
__global__ void __launch_bounds__(256)
knn_kernel(const float* __restrict__ xyz, int* __restrict__ idxout) {
  __shared__ __align__(16) float sxyz[NN * 3];
  const int batch = blockIdx.x >> 7;          // S/8 = 128 groups per batch
  const int sgrp  = blockIdx.x & 127;
  const float4* src = (const float4*)(xyz + (size_t)batch * NN * 3);
  float4* dst = (float4*)sxyz;
  for (int i = threadIdx.x; i < NN * 3 / 4; i += blockDim.x) dst[i] = src[i];
  __syncthreads();

  const int wave = threadIdx.x >> 5, lane = threadIdx.x & 31;
  const int s = sgrp * 8 + wave;
  const float qx = sxyz[s * 3 + 0], qy = sxyz[s * 3 + 1], qz = sxyz[s * 3 + 2];
  int* op = idxout + ((size_t)batch * SS + s) * KK;

  unsigned long long last = 0ull;
  for (int it = 0; it < KK; ++it) {
    unsigned long long best = ~0ull;
    for (int p = lane; p < NN; p += 32) {
      float dx = sxyz[p * 3 + 0] - qx;
      float dy = sxyz[p * 3 + 1] - qy;
      float dz = sxyz[p * 3 + 2] - qz;
      float d  = fmaf(dx, dx, fmaf(dy, dy, dz * dz));
      unsigned long long key =
          ((unsigned long long)__float_as_uint(d) << 32) | (unsigned)p;
      bool ok = (it == 0) || (key > last);
      if (ok && key < best) best = key;
    }
    #pragma unroll
    for (int m = 16; m; m >>= 1) {
      unsigned hi = __shfl_xor((unsigned)(best >> 32), m, 32);
      unsigned lo = __shfl_xor((unsigned)best, m, 32);
      unsigned long long o = ((unsigned long long)hi << 32) | lo;
      if (o < best) best = o;
    }
    last = best;
    if (lane == 0) op[it] = (int)(best & 0xffffffffu);
  }
}

// ------------------------------- gather ------------------------------------
// Build feats0 (MROWS x 96, bf16): [dxyz(3) | points(64) | zero pad(29)].
// One wave per 8 rows; lane=channel for coalesced point loads/stores.
__global__ void __launch_bounds__(256)
gather_kernel(const float* __restrict__ xyz, const float* __restrict__ pts,
              const int* __restrict__ idx, __bf16* __restrict__ feats) {
  const int wave = threadIdx.x >> 5, lane = threadIdx.x & 31;
  const size_t row0 = ((size_t)blockIdx.x * 8 + wave) * 8;
  for (int rr = 0; rr < 8; ++rr) {
    size_t row = row0 + rr;                    // row = ((b*S+s)*K + k)
    size_t bs  = row >> 5;
    int s = (int)(bs & (SS - 1));
    int b = (int)(bs >> 10);
    int j = idx[row];
    const float* pj = pts + ((size_t)b * NN + j) * DF;
    __bf16* fr = feats + row * C0PAD;
    // points channels: lane and lane+32 -> feats channels 3+lane, 35+lane
    fr[3 + lane]  = (__bf16)pj[lane];
    fr[35 + lane] = (__bf16)pj[lane + 32];
    if (lane < 3) {
      float d = xyz[((size_t)b * NN + j) * 3 + lane] -
                xyz[((size_t)b * NN + s) * 3 + lane];
      fr[lane] = (__bf16)d;
    }
    if (lane < 29) fr[67 + lane] = (__bf16)0.0f;   // pad 67..95
  }
}

// ------------------------------- WMMA GEMM ---------------------------------
// D = A(Mx32k) x W^T + bias, optional fused BN+ReLU on A-load (prev layer's
// affine a*h+b), bf16 storage of H, fused per-channel sum/sumsq statistics.
// Wave handles 16 rows x full Cout (= NT*16). Block = 8 waves = 128 rows.
template <int KCHUNKS, int NT, bool BN>
__global__ void __launch_bounds__(256)
gemm_bn_kernel(const __bf16* __restrict__ A, int Apad,
               const __bf16* __restrict__ Wb, int Wpad,
               const float* __restrict__ bias,
               const float* __restrict__ bnA, const float* __restrict__ bnB,
               __bf16* __restrict__ H,
               float* __restrict__ gsum, float* __restrict__ gsq) {
  __shared__ float ssum[NT * 16];
  __shared__ float ssq[NT * 16];
  const int tid = threadIdx.x;
  if (tid < NT * 16) { ssum[tid] = 0.0f; ssq[tid] = 0.0f; }
  __syncthreads();

  const int wave = tid >> 5, lane = tid & 31;
  const int lhalf = lane >> 4;          // 0: lanes 0-15, 1: lanes 16-31
  const int lcol  = lane & 15;
  const int row0  = (blockIdx.x * 8 + wave) * 16;

  v8f acc[NT];
  #pragma unroll
  for (int nt = 0; nt < NT; ++nt) {
    float bv = bias[nt * 16 + lcol];
    #pragma unroll
    for (int r = 0; r < 8; ++r) acc[nt][r] = bv;
  }

  #pragma unroll
  for (int kc = 0; kc < KCHUNKS; ++kc) {
    // ---- A fragment: 16x32 bf16, ISA layout. m = lcol (per half-wave). ----
    const int k0 = kc * 32 + (lhalf ? 8 : 0);
    const __bf16* ap = A + (size_t)(row0 + lcol) * Apad + k0;
    v8bf lo = *(const v8bf*)ap;            // K = k0 .. k0+7
    v8bf hi = *(const v8bf*)(ap + 16);     // K = k0+16 .. k0+23
    v16bf af;
    if (BN) {
      #pragma unroll
      for (int i = 0; i < 8; ++i) {
        float x = fmaf(bnA[k0 + i],      (float)lo[i], bnB[k0 + i]);
        float y = fmaf(bnA[k0 + 16 + i], (float)hi[i], bnB[k0 + 16 + i]);
        af[i]     = (__bf16)fmaxf(x, 0.0f);
        af[i + 8] = (__bf16)fmaxf(y, 0.0f);
      }
    } else {
      #pragma unroll
      for (int i = 0; i < 8; ++i) { af[i] = lo[i]; af[i + 8] = hi[i]; }
    }
    // ---- B fragments per N-tile: column o = nt*16+lcol, same K order as A. ----
    #pragma unroll
    for (int nt = 0; nt < NT; ++nt) {
      const int ocol = nt * 16 + lcol;
      const __bf16* wp = Wb + (size_t)ocol * Wpad + kc * 32 + (lhalf ? 8 : 0);
      v8bf wlo = *(const v8bf*)wp;         // K = base .. base+7
      v8bf whi = *(const v8bf*)(wp + 16);  // K = base+16 .. base+23
      v16bf bf;
      #pragma unroll
      for (int i = 0; i < 8; ++i) { bf[i] = wlo[i]; bf[i + 8] = whi[i]; }
      acc[nt] = __builtin_amdgcn_wmma_f32_16x16x32_bf16(
          false, af, false, bf, (short)0, acc[nt], false, false);
    }
  }

  // ---- epilogue: bf16 store of H, fused BN statistics. ----
  #pragma unroll
  for (int nt = 0; nt < NT; ++nt) {
    const int ocol = nt * 16 + lcol;
    float s = 0.0f, s2 = 0.0f;
    #pragma unroll
    for (int r = 0; r < 8; ++r) {
      float v = acc[nt][r];
      int rrow = row0 + r + lhalf * 8;     // D layout: M=r / M=r+8 per half
      H[(size_t)rrow * (NT * 16) + ocol] = (__bf16)v;
      s += v; s2 += v * v;
    }
    atomicAdd(&ssum[ocol], s);             // ds_add_f32
    atomicAdd(&ssq[ocol], s2);
  }
  __syncthreads();
  if (tid < NT * 16) {
    atomicAdd(&gsum[tid], ssum[tid]);      // global_atomic_add_f32
    atomicAdd(&gsq[tid],  ssq[tid]);
  }
}

// ---------------------- BN finalize: a = g*rsqrt(var+eps), b = be - mean*a --
__global__ void bn_finalize_kernel(const float* __restrict__ sum,
                                   const float* __restrict__ sq,
                                   const float* __restrict__ g,
                                   const float* __restrict__ be,
                                   float* __restrict__ a, float* __restrict__ b,
                                   int C, float invM) {
  int c = threadIdx.x;
  if (c >= C) return;
  float mean = sum[c] * invM;
  float var  = sq[c] * invM - mean * mean;
  float inv  = rsqrtf(var + 1e-5f);
  float ac   = g[c] * inv;
  a[c] = ac;
  b[c] = be[c] - mean * ac;
}

// ---------------------- maxpool over K with fused BN+ReLU -------------------
__global__ void __launch_bounds__(128)
maxpool_kernel(const __bf16* __restrict__ H2, const float* __restrict__ a,
               const float* __restrict__ b, float* __restrict__ out_pts) {
  const int bs = blockIdx.x;               // 0 .. B*S-1
  const int c  = threadIdx.x;              // 0 .. 127
  const __bf16* base = H2 + (size_t)bs * KK * 128 + c;
  const float ac = a[c], bc = b[c];
  float m = 0.0f;                          // relu(...) >= 0
  #pragma unroll 4
  for (int k = 0; k < KK; ++k) {
    float v = fmaxf(fmaf(ac, (float)base[(size_t)k * 128], bc), 0.0f);
    m = fmaxf(m, v);
  }
  out_pts[(size_t)bs * 128 + c] = m;
}

// ---------------------- new_xyz = xyz[:, :S, :] -----------------------------
__global__ void copy_xyz_kernel(const float* __restrict__ xyz,
                                float* __restrict__ out, int total) {
  int i = blockIdx.x * blockDim.x + threadIdx.x;
  if (i >= total) return;
  int b = i / (SS * 3);
  int r = i - b * (SS * 3);
  out[i] = xyz[(size_t)b * NN * 3 + r];
}

// ---------------------------------------------------------------------------
extern "C" void kernel_launch(void* const* d_in, const int* in_sizes, int n_in,
                              void* d_out, int out_size, void* d_ws, size_t ws_size,
                              hipStream_t stream) {
  const float* xyz = (const float*)d_in[0];
  const float* pts = (const float*)d_in[1];
  const float* W0  = (const float*)d_in[2];
  const float* b0  = (const float*)d_in[3];
  const float* g0  = (const float*)d_in[4];
  const float* be0 = (const float*)d_in[5];
  const float* W1  = (const float*)d_in[6];
  const float* b1  = (const float*)d_in[7];
  const float* g1  = (const float*)d_in[8];
  const float* be1 = (const float*)d_in[9];
  const float* W2  = (const float*)d_in[10];
  const float* b2  = (const float*)d_in[11];
  const float* g2  = (const float*)d_in[12];
  const float* be2 = (const float*)d_in[13];
  float* out = (float*)d_out;
  float* out_xyz = out;                       // B*S*3
  float* out_pts = out + (size_t)BB * SS * 3; // B*S*128

  // ---- workspace carve-up (256B aligned) ----
  char* w = (char*)d_ws;
  auto alloc = [&](size_t bytes) -> char* {
    char* p = w; w += (bytes + 255) & ~(size_t)255; return p;
  };
  int*    idxbuf = (int*)   alloc((size_t)MROWS * sizeof(int));
  __bf16* feats0 = (__bf16*)alloc((size_t)MROWS * C0PAD * sizeof(__bf16));
  __bf16* h0     = (__bf16*)alloc((size_t)MROWS * 64  * sizeof(__bf16));
  __bf16* h1     = (__bf16*)alloc((size_t)MROWS * 64  * sizeof(__bf16));
  __bf16* h2     = (__bf16*)alloc((size_t)MROWS * 128 * sizeof(__bf16));
  __bf16* w0bf   = (__bf16*)alloc((size_t)64  * C0PAD * sizeof(__bf16));
  __bf16* w1bf   = (__bf16*)alloc((size_t)64  * 64    * sizeof(__bf16));
  __bf16* w2bf   = (__bf16*)alloc((size_t)128 * 64    * sizeof(__bf16));
  float*  stats  = (float*) alloc(6 * 128 * sizeof(float));  // sum/sq x3
  float*  abuf   = (float*) alloc(6 * 128 * sizeof(float));  // a/b x3
  float* sum0 = stats,        *sq0 = stats + 128;
  float* sum1 = stats + 256,  *sq1 = stats + 384;
  float* sum2 = stats + 512,  *sq2 = stats + 640;
  float* a0 = abuf,        *bb0 = abuf + 128;
  float* a1 = abuf + 256,  *bb1 = abuf + 384;
  float* a2 = abuf + 512,  *bb2 = abuf + 640;

  const float invM = 1.0f / (float)MROWS;

  // 0) zero the BN statistic accumulators (deterministic per call)
  zero_f32_kernel<<<3, 256, 0, stream>>>(stats, 6 * 128);

  // 1) KNN: 2048 blocks x 8 waves (one wave per query)
  knn_kernel<<<BB * (SS / 8), 256, 0, stream>>>(xyz, idxbuf);

  // 2) gather + concat -> bf16 feats (padded to 96 channels)
  gather_kernel<<<MROWS / 64, 256, 0, stream>>>(xyz, pts, idxbuf, feats0);

  // 3) weight prep (f32 -> bf16, K-padded)
  prep_w_kernel<<<(64 * C0PAD + 255) / 256, 256, 0, stream>>>(W0, w0bf, 64, 67, C0PAD);
  prep_w_kernel<<<(64 * 64    + 255) / 256, 256, 0, stream>>>(W1, w1bf, 64, 64, 64);
  prep_w_kernel<<<(128 * 64   + 255) / 256, 256, 0, stream>>>(W2, w2bf, 128, 64, 64);

  const int gemm_blocks = MROWS / 128;   // 8 waves x 16 rows per block

  // 4) layer 0: feats0 @ W0^T + b0  (no BN on input), stats fused
  gemm_bn_kernel<3, 4, false><<<gemm_blocks, 256, 0, stream>>>(
      feats0, C0PAD, w0bf, C0PAD, b0, nullptr, nullptr, h0, sum0, sq0);
  bn_finalize_kernel<<<1, 128, 0, stream>>>(sum0, sq0, g0, be0, a0, bb0, 64, invM);

  // 5) layer 1: BN0+ReLU fused into A-load
  gemm_bn_kernel<2, 4, true><<<gemm_blocks, 256, 0, stream>>>(
      h0, 64, w1bf, 64, b1, a0, bb0, h1, sum1, sq1);
  bn_finalize_kernel<<<1, 128, 0, stream>>>(sum1, sq1, g1, be1, a1, bb1, 64, invM);

  // 6) layer 2: BN1+ReLU fused into A-load, Cout=128
  gemm_bn_kernel<2, 8, true><<<gemm_blocks, 256, 0, stream>>>(
      h1, 64, w2bf, 64, b2, a1, bb1, h2, sum2, sq2);
  bn_finalize_kernel<<<1, 128, 0, stream>>>(sum2, sq2, g2, be2, a2, bb2, 128, invM);

  // 7) max over K with fused BN2+ReLU -> new_points (f32)
  maxpool_kernel<<<BB * SS, 128, 0, stream>>>(h2, a2, bb2, out_pts);

  // 8) new_xyz passthrough
  copy_xyz_kernel<<<(BB * SS * 3 + 255) / 256, 256, 0, stream>>>(
      xyz, out_xyz, BB * SS * 3);
}
